// HardMoeBlock_35089882808653
// MI455X (gfx1250) — matrix operand order repaired and checked by reading, not compiled, not run on hardware
//
#include <hip/hip_runtime.h>
#include <hip/hip_bf16.h>
#include <cstdint>
#include <cstddef>

// Problem constants (B=2, T=1024, D=1024, H=4096, K=8)
#define NT     2048
#define D_DIM  1024
#define H_DIM  4096
#define NEXP   8

typedef __attribute__((ext_vector_type(16))) __bf16        v16bf;
typedef __attribute__((ext_vector_type(8)))  float         v8f;
typedef __attribute__((ext_vector_type(4)))  unsigned int  v4u;
typedef __attribute__((ext_vector_type(4)))  float         v4f;
typedef __attribute__((ext_vector_type(4)))  __bf16        v4bf;
typedef __attribute__((ext_vector_type(2)))  float         v2f;
typedef __attribute__((ext_vector_type(2)))  __bf16        v2bf;

union FragB16 { v4u q[2]; v16bf v; };   // 16 bf16 = 8 VGPRs

__device__ __forceinline__ v8f vzero8() {
  v8f z;
#pragma unroll
  for (int i = 0; i < 8; ++i) z[i] = 0.0f;
  return z;
}

__device__ __forceinline__ unsigned pack_bf2(float lo, float hi) {
  v2f t; t[0] = lo; t[1] = hi;
  return __builtin_bit_cast(unsigned, __builtin_convertvector(t, v2bf));
}

// ---------------------------------------------------------------------------
// Kernel 1: routing. One block. argmax over one-hot combine weights, count,
// exclusive prefix, compacted perm (slot -> token). Deterministic output.
// ---------------------------------------------------------------------------
__global__ void route_kernel(const float* __restrict__ cw,
                             int* __restrict__ cnt, int* __restrict__ off,
                             int* __restrict__ perm) {
  __shared__ int lcnt[NEXP];
  __shared__ int lcur[NEXP];
  const int tid = threadIdx.x;
  if (tid < NEXP) lcnt[tid] = 0;
  __syncthreads();

  int myexp[NT / 256];
#pragma unroll
  for (int i = 0; i < NT / 256; ++i) {
    const int t = tid + i * 256;
    const float* w = cw + (size_t)t * NEXP;
    int best = 0; float bw = w[0];
#pragma unroll
    for (int k = 1; k < NEXP; ++k) { float v = w[k]; if (v > bw) { bw = v; best = k; } }
    myexp[i] = best;
    atomicAdd(&lcnt[best], 1);
  }
  __syncthreads();
  if (tid == 0) {
    int acc = 0;
    for (int k = 0; k < NEXP; ++k) {
      off[k] = acc; lcur[k] = acc; cnt[k] = lcnt[k]; acc += lcnt[k];
    }
  }
  __syncthreads();
#pragma unroll
  for (int i = 0; i < NT / 256; ++i) {
    const int t = tid + i * 256;
    const int slot = atomicAdd(&lcur[myexp[i]], 1);
    perm[slot] = t;
  }
}

// ---------------------------------------------------------------------------
// Kernel 2: x fp32 -> bf16 (native packed converts)
// ---------------------------------------------------------------------------
__global__ void cvt_kernel(const float* __restrict__ x,
                           __bf16* __restrict__ xb) {
  const int i = blockIdx.x * blockDim.x + threadIdx.x;   // * 4 elements
  const v4f v = ((const v4f*)x)[i];
  ((v4bf*)xb)[i] = __builtin_convertvector(v, v4bf);
}

// ---------------------------------------------------------------------------
// Staging: a 32x16 fp32 weight slice is loaded as: lane (rp = lane&15,
// ch = lane>>4) owns K-rows {2rp, 2rp+1} x 8 columns [8*ch, 8*ch+8).
// B-fragment order in LDS (lanes 0-15: K 0..7,16..23, lanes 16-31: K 8..15,24..31,
// element = (K & 7) + 8*(K >> 4)) puts rows (2rp, 2rp+1) at adjacent bf16 elements of the
// same destination lane -> one v_cvt_pk_bf16_f32 + one ds_store_b32 per col.
// ---------------------------------------------------------------------------
__device__ __forceinline__ void load_rows2(const float* __restrict__ W,
                                           size_t ld, int k0, int rp, int ch,
                                           int col0, float4 r[4]) {
  const float4* p0 = (const float4*)(W + (size_t)(k0 + 2 * rp)     * ld + col0 + 8 * ch);
  const float4* p1 = (const float4*)(W + (size_t)(k0 + 2 * rp + 1) * ld + col0 + 8 * ch);
  r[0] = p0[0]; r[1] = p0[1];
  r[2] = p1[0]; r[3] = p1[1];
}

__device__ __forceinline__ void stage_store2(__bf16* __restrict__ s,
                                             const float4 r[4],
                                             int rp, int ch) {
  unsigned* s32 = (unsigned*)s;
  // u32 index = ((hh + (((rp>>2)&1) ? 16 : 0)) * 16 + 2*(rp&3) + (rp>=8 ? 8 : 0)) / 2, hh = 8*ch + c
  const int base = 64 * ch + (((rp >> 2) & 1) ? 128 : 0) + (rp & 3) + ((rp >= 8) ? 4 : 0);
  const float lo[8] = {r[0].x, r[0].y, r[0].z, r[0].w,
                       r[1].x, r[1].y, r[1].z, r[1].w};
  const float hi[8] = {r[2].x, r[2].y, r[2].z, r[2].w,
                       r[3].x, r[3].y, r[3].z, r[3].w};
#pragma unroll
  for (int c = 0; c < 8; ++c)
    s32[base + c * 8] = pack_bf2(lo[c], hi[c]);
}

__device__ __forceinline__ void load_bfrag(const __bf16* __restrict__ s,
                                           int lane, FragB16& f) {
  const v4u* pb = (const v4u*)(s + lane * 16);
  f.q[0] = pb[0]; f.q[1] = pb[1];
}

__device__ __forceinline__ void load_afrag(const __bf16* __restrict__ src,
                                           FragB16& f) {
  const v4u* pa = (const v4u*)src;
  f.q[0] = pa[0];        // K base..base+7
  f.q[1] = pa[2];        // K base+16..base+23
}

// ---------------------------------------------------------------------------
// Kernel 3: gate/up GEMMs + SiLU, per expert group.
// Block = 128 thr (4 waves). Tile: 64 tokens x 64 H-cols (wave owns 16 cols).
// Double-buffered LDS staging behind the WMMAs.
// ---------------------------------------------------------------------------
__global__ void __launch_bounds__(128)
moe_gateup_kernel(const __bf16* __restrict__ xb,
                  const float* __restrict__ wg, const float* __restrict__ wu,
                  const int* __restrict__ cnt, const int* __restrict__ off,
                  __bf16* __restrict__ hb) {
  const int e = blockIdx.z;
  const int n = cnt[e];
  const int rowblk = blockIdx.y * 64;
  if (rowblk >= n) return;                 // block-uniform exit
  const int base = off[e];
  const int wave = threadIdx.x >> 5;
  const int lane = threadIdx.x & 31;
  const int h0 = blockIdx.x * 64 + wave * 16;
  const size_t we = (size_t)e * D_DIM * H_DIM;
  const float* Wg = wg + we;
  const float* Wu = wu + we;

  __shared__ alignas(16) __bf16 sB[2][4][2][512];   // [buf][wave][g/u][...]

  v8f accg[4], accu[4];
#pragma unroll
  for (int r = 0; r < 4; ++r) { accg[r] = vzero8(); accu[r] = vzero8(); }

  int arow[4];
#pragma unroll
  for (int r = 0; r < 4; ++r) {
    int s = rowblk + r * 16 + (lane & 15);
    if (s >= n) s = 0;                     // dummy row, never written back
    arow[r] = base + s;
  }
  const int kbA = (lane < 16) ? 0 : 8;
  const int rp  = lane & 15;
  const int ch  = lane >> 4;

  float4 rg[4], ru[4];
  // Prologue: stage K-step 0 into buffer 0
  load_rows2(Wg, H_DIM, 0, rp, ch, h0, rg);
  load_rows2(Wu, H_DIM, 0, rp, ch, h0, ru);
  stage_store2(&sB[0][wave][0][0], rg, rp, ch);
  stage_store2(&sB[0][wave][1][0], ru, rp, ch);

  int p = 0;
  for (int k0 = 0; k0 < D_DIM; k0 += 32, p ^= 1) {
    const bool more = (k0 + 32) < D_DIM;
    if (more) {                             // issue next tile's global loads
      load_rows2(Wg, H_DIM, k0 + 32, rp, ch, h0, rg);
      load_rows2(Wu, H_DIM, k0 + 32, rp, ch, h0, ru);
    }
    FragB16 fg, fu;
    load_bfrag(&sB[p][wave][0][0], lane, fg);
    load_bfrag(&sB[p][wave][1][0], lane, fu);
    FragB16 fa[4];
#pragma unroll
    for (int r = 0; r < 4; ++r)
      load_afrag(xb + (size_t)arow[r] * D_DIM + k0 + kbA, fa[r]);
#pragma unroll
    for (int r = 0; r < 4; ++r) {
      accg[r] = __builtin_amdgcn_wmma_f32_16x16x32_bf16(
          false, fa[r].v, false, fg.v, (short)0, accg[r], false, false);
      accu[r] = __builtin_amdgcn_wmma_f32_16x16x32_bf16(
          false, fa[r].v, false, fu.v, (short)0, accu[r], false, false);
    }
    if (more) {                             // stage next tile behind the WMMAs
      stage_store2(&sB[p ^ 1][wave][0][0], rg, rp, ch);
      stage_store2(&sB[p ^ 1][wave][1][0], ru, rp, ch);
    }
  }

  // Epilogue: h = silu(g) * u = g * rcp(1+exp(-g)) * u, bf16 store.
  const int col = h0 + (lane & 15);
  const int rh  = (lane < 16) ? 0 : 8;     // C layout: VGPR j -> row j / j+8
#pragma unroll
  for (int r = 0; r < 4; ++r) {
#pragma unroll
    for (int j = 0; j < 8; ++j) {
      const int s = rowblk + r * 16 + j + rh;
      if (s < n) {
        const float g = accg[r][j];
        const float u = accu[r][j];
        const float sg = __builtin_amdgcn_rcpf(1.0f + __expf(-g));
        hb[(size_t)(base + s) * H_DIM + col] = (__bf16)(g * sg * u);
      }
    }
  }
}

// ---------------------------------------------------------------------------
// Kernel 4: down projection y = h @ wd, scaled by combine weight, scattered
// back to token order. Same pipelined tiling (K runs over H=4096).
// ---------------------------------------------------------------------------
__global__ void __launch_bounds__(128)
moe_down_kernel(const __bf16* __restrict__ hb,
                const float* __restrict__ wd,
                const int* __restrict__ cnt, const int* __restrict__ off,
                const int* __restrict__ perm, const float* __restrict__ cw,
                float* __restrict__ out) {
  const int e = blockIdx.z;
  const int n = cnt[e];
  const int rowblk = blockIdx.y * 64;
  if (rowblk >= n) return;
  const int base = off[e];
  const int wave = threadIdx.x >> 5;
  const int lane = threadIdx.x & 31;
  const int d0 = blockIdx.x * 64 + wave * 16;
  const float* Wd = wd + (size_t)e * H_DIM * D_DIM;

  __shared__ alignas(16) __bf16 sB[2][4][512];

  v8f acc[4];
#pragma unroll
  for (int r = 0; r < 4; ++r) acc[r] = vzero8();

  int arow[4];
#pragma unroll
  for (int r = 0; r < 4; ++r) {
    int s = rowblk + r * 16 + (lane & 15);
    if (s >= n) s = 0;
    arow[r] = base + s;
  }
  const int kbA = (lane < 16) ? 0 : 8;
  const int rp  = lane & 15;
  const int ch  = lane >> 4;

  float4 rd[4];
  load_rows2(Wd, D_DIM, 0, rp, ch, d0, rd);
  stage_store2(&sB[0][wave][0], rd, rp, ch);

  int p = 0;
  for (int k0 = 0; k0 < H_DIM; k0 += 32, p ^= 1) {
    const bool more = (k0 + 32) < H_DIM;
    if (more) load_rows2(Wd, D_DIM, k0 + 32, rp, ch, d0, rd);
    FragB16 fb;
    load_bfrag(&sB[p][wave][0], lane, fb);
    FragB16 fa[4];
#pragma unroll
    for (int r = 0; r < 4; ++r)
      load_afrag(hb + (size_t)arow[r] * H_DIM + k0 + kbA, fa[r]);
#pragma unroll
    for (int r = 0; r < 4; ++r) {
      acc[r] = __builtin_amdgcn_wmma_f32_16x16x32_bf16(
          false, fa[r].v, false, fb.v, (short)0, acc[r], false, false);
    }
    if (more) stage_store2(&sB[p ^ 1][wave][0], rd, rp, ch);
  }

  const int col = d0 + (lane & 15);
  const int rh  = (lane < 16) ? 0 : 8;
#pragma unroll
  for (int r = 0; r < 4; ++r) {
#pragma unroll
    for (int j = 0; j < 8; ++j) {
      const int s = rowblk + r * 16 + j + rh;
      if (s < n) {
        const int tok = perm[base + s];
        const float w = cw[(size_t)tok * NEXP + e];   // == 1.0 (one-hot)
        out[(size_t)tok * D_DIM + col] = acc[r][j] * w;
      }
    }
  }
}

// ---------------------------------------------------------------------------
// Launch: route -> cvt(x) -> gate/up -> down.  ws layout (~21 MB):
//   [0,32)      int cnt[8]
//   [32,64)     int off[8]
//   [64,8256)   int perm[2048]
//   [16384, +4MiB)   x as bf16   [2048 x 1024]
//   [+4MiB, +20MiB)  h as bf16   [2048 x 4096]
// ---------------------------------------------------------------------------
extern "C" void kernel_launch(void* const* d_in, const int* in_sizes, int n_in,
                              void* d_out, int out_size, void* d_ws, size_t ws_size,
                              hipStream_t stream) {
  (void)in_sizes; (void)n_in; (void)out_size; (void)ws_size;
  const float* x  = (const float*)d_in[0];
  const float* cw = (const float*)d_in[1];
  const float* wg = (const float*)d_in[2];
  const float* wu = (const float*)d_in[3];
  const float* wd = (const float*)d_in[4];
  float* out = (float*)d_out;

  char* ws = (char*)d_ws;
  int*  cnt  = (int*)(ws + 0);
  int*  off  = (int*)(ws + 32);
  int*  perm = (int*)(ws + 64);
  __bf16* xb = (__bf16*)(ws + 16384);
  __bf16* hb = (__bf16*)(ws + 16384 + (size_t)NT * D_DIM * 2);

  route_kernel<<<1, 256, 0, stream>>>(cw, cnt, off, perm);

  // NT*D_DIM / (4 per thread) / 256 threads = 2048 blocks
  cvt_kernel<<<(NT * D_DIM) / (4 * 256), 256, 0, stream>>>(x, xb);

  moe_gateup_kernel<<<dim3(H_DIM / 64, NT / 64, NEXP), 128, 0, stream>>>(
      xb, wg, wu, cnt, off, hb);

  moe_down_kernel<<<dim3(D_DIM / 64, NT / 64, NEXP), 128, 0, stream>>>(
      hb, wd, cnt, off, perm, cw, out);
}
